// MoBA_86646670229621
// MI455X (gfx1250) — matrix operand order repaired and checked by reading, not compiled, not run on hardware
//
#include <hip/hip_runtime.h>
#include <hip/hip_bf16.h>

typedef __bf16 bf16;
typedef __attribute__((ext_vector_type(16))) __bf16 v16bf;
typedef __attribute__((ext_vector_type(8)))  __bf16 v8bf;
typedef __attribute__((ext_vector_type(8)))  float  v8f;

#define B_ 8
#define S_ 4096
#define D_ 768
#define E_ 8
#define R_ 8
#define NTOK   (B_*S_)      // 32768 tokens
#define NTILES (NTOK/16)    // 2048 tiles of 16 tokens
#define KSTEPS (D_/32)      // 24 bf16 WMMA k-steps over D
#define NT_UP  (D_/16)      // 48 output N-tiles
#define KUP    3            // 96 / 32 (64 hg rows + 8 gate rows + 24 zero pad)
#define SCR_STRIDE 104      // 96 cols + 8 pad (bank spread)

#define WD_E (4*KSTEPS*32*16)      // packed Wd fragments   (96 KB bf16)
#define WG_E (KSTEPS*32*16)        // packed Wg fragments   (24 KB)
#define WU_E (NT_UP*KUP*32*16)     // packed W' fragments   (144 KB)
#define SMEM_BYTES (WD_E*2 + WG_E*2 + WU_E*2 + (512+64+64+16)*4 + 8*16*SCR_STRIDE*2)

__device__ __forceinline__ bf16 f2bf(float f) {           // RNE f32 -> bf16 (cold paths)
  union { float f; unsigned u; } v; v.f = f;
  unsigned r = v.u + 0x7FFFu + ((v.u >> 16) & 1u);
  unsigned short h = (unsigned short)(r >> 16);
  return __builtin_bit_cast(bf16, h);
}

__device__ __forceinline__ v8f wmma_bf16(v16bf a, v16bf b, v8f c) {
  return __builtin_amdgcn_wmma_f32_16x16x32_bf16(false, a, false, b, (short)0, c, false, false);
}

__device__ __forceinline__ v16bf ldsB(const bf16* p) { return *(const v16bf*)p; }

// A fragment of x (f32 global -> bf16 regs, packed converts). Contiguous A layout:
// lanes 0-15: K = k0+{0..15}; lanes 16-31: K = k0+{16..31}; M = lane&15.
__device__ __forceinline__ v16bf load_a_global(const float* xrow, int kbase, int hi) {
  int o1 = kbase + (hi ? 16 : 0);
  int o2 = kbase + (hi ? 24 : 8);
  v8f f0 = *(const v8f*)(xrow + o1);   // 32B-aligned (o1 % 8 == 0)
  v8f f1 = *(const v8f*)(xrow + o2);
  v8bf b0 = __builtin_convertvector(f0, v8bf);   // v_cvt_pk_bf16_f32 path
  v8bf b1 = __builtin_convertvector(f1, v8bf);
  return __builtin_shufflevector(b0, b1, 0,1,2,3,4,5,6,7,8,9,10,11,12,13,14,15);
}

// A' fragment from per-wave LDS scratch (two 16B ds_load_b128 runs)
__device__ __forceinline__ v16bf load_a_scratch(const bf16* scr_row, int kk, int hi) {
  int o1 = kk*32 + (hi ? 16 : 0);
  int o2 = kk*32 + (hi ? 24 : 8);
  v8bf r0 = *(const v8bf*)(scr_row + o1);
  v8bf r1 = *(const v8bf*)(scr_row + o2);
  return __builtin_shufflevector(r0, r1, 0,1,2,3,4,5,6,7,8,9,10,11,12,13,14,15);
}

extern "C" __global__ __launch_bounds__(256, 1)
void moe_fused(const float* __restrict__ x,  const float* __restrict__ Wd,
               const float* __restrict__ bd, const float* __restrict__ Wm,
               const float* __restrict__ bm, const float* __restrict__ Wu,
               const float* __restrict__ bu, const float* __restrict__ Wg,
               const float* __restrict__ bg, float* __restrict__ out)
{
  extern __shared__ char smem[];
  bf16*  sWd = (bf16*)smem;            // [chain4][ks24][lane32][16]
  bf16*  sWg = sWd + WD_E;             // [ks24][lane32][16]
  bf16*  sWu = sWg + WG_E;             // [nt48][kk3][lane32][16]
  float* sWm = (float*)(sWu + WU_E);   // [8][8][8]
  float* sbd = sWm + 512;
  float* sbm = sbd + 64;
  float* sbg = sbm + 64;               // 16 floats (padded)
  bf16*  sScr = (bf16*)(sbg + 16);     // [wave8][16][SCR_STRIDE]

  const int tid = threadIdx.x;

  // ---- stage weights (bf16, pre-swizzled into B-fragment order) ----
  for (int i = tid; i < WD_E; i += 256) {               // Wd: 2 experts per N16 tile
    int j = i & 15, lane = (i >> 4) & 31, g = i >> 9;
    int ks = g % KSTEPS, c = g / KSTEPS;
    int k = ks*32 + (lane < 16 ? j : 16 + j);
    int n = lane & 15, e = c*2 + (n >> 3), r = n & 7;
    sWd[i] = f2bf(Wd[(e*D_ + k)*R_ + r]);
  }
  for (int i = tid; i < WG_E; i += 256) {               // gate chain (cols 8..15 = 0)
    int j = i & 15, lane = (i >> 4) & 31, ks = i >> 9;
    int k = ks*32 + (lane < 16 ? j : 16 + j);
    int n = lane & 15;
    sWg[i] = (n < E_) ? f2bf(Wg[k*E_ + n]) : f2bf(0.0f);
  }
  for (int i = tid; i < WU_E; i += 256) {               // W' = [Wu(64); bu(8); 0(24)]
    int j = i & 15, lane = (i >> 4) & 31, g = i >> 9;
    int kk = g % KUP, nt = g / KUP;
    int k = kk*32 + (lane < 16 ? j : 16 + j);
    int n = nt*16 + (lane & 15);
    float v = 0.0f;
    if (k < 64)      v = Wu[((k >> 3)*R_ + (k & 7))*D_ + n];
    else if (k < 72) v = bu[(k - 64)*D_ + n];
    sWu[i] = f2bf(v);
  }
  for (int i = tid; i < 512; i += 256) sWm[i] = Wm[i];
  if (tid < 64) sbd[tid] = bd[tid];
  if (tid < 64) sbm[tid] = bm[tid];
  if (tid < E_) sbg[tid] = bg[tid];
  __syncthreads();

  const int wave = tid >> 5, lane = tid & 31;
  const int li = lane & 15, hi = (lane >> 4) & 1, q = lane & 7;
  bf16* scr = sScr + wave * 16 * SCR_STRIDE;

  // zero the K-pad columns (72..95) of per-wave scratch once
  for (int i = lane; i < 16*24; i += 32)
    scr[(i / 24)*SCR_STRIDE + 72 + (i % 24)] = f2bf(0.0f);

  const int nwaves = gridDim.x * 8;
  for (int t = blockIdx.x * 8 + wave; t < NTILES; t += nwaves) {
    const int m0 = t * 16;
    const float* xrow = x + (size_t)(m0 + li) * D_;

    // ---- down projection (4 expert chains) + gating logits, K = 768 ----
    v8f ag = {0,0,0,0,0,0,0,0};
    v8f accd[4];
    #pragma unroll
    for (int c = 0; c < 4; ++c) accd[c] = (v8f){0,0,0,0,0,0,0,0};
    for (int ks = 0; ks < KSTEPS; ++ks) {
      v16bf a = load_a_global(xrow, ks*32, hi);
      ag = wmma_bf16(a, ldsB(sWg + (ks*32 + lane)*16), ag);
      #pragma unroll
      for (int c = 0; c < 4; ++c)
        accd[c] = wmma_bf16(a, ldsB(sWd + ((c*KSTEPS + ks)*32 + lane)*16), accd[c]);
    }

    // ---- softmax over experts (width-8 lane butterflies; cols 0..7 valid) ----
    float bgv = sbg[q];
    v8f gate;
    #pragma unroll
    for (int v = 0; v < 8; ++v) {
      float l = ag[v] + bgv;
      float mx = l;
      mx = fmaxf(mx, __shfl_xor(mx, 1, 8));
      mx = fmaxf(mx, __shfl_xor(mx, 2, 8));
      mx = fmaxf(mx, __shfl_xor(mx, 4, 8));
      float ex = __expf(l - mx);
      float sm = ex;
      sm += __shfl_xor(sm, 1, 8);
      sm += __shfl_xor(sm, 2, 8);
      sm += __shfl_xor(sm, 4, 8);
      gate[v] = ex / sm;
    }

    // ---- mid 8x8 per-expert GEMM + biases + gate fold; transpose to A' in LDS ----
    #pragma unroll
    for (int c = 0; c < 4; ++c) {
      int e = c*2 + (li >> 3);
      float bdv = sbd[e*8 + q];
      float bmv = sbm[e*8 + q];
      // Wm scalars pre-loaded in xor order: wx[m] = Wm[e][q^m][q]
      // (constant index m -> no register-indexed selects)
      float wx[8];
      #pragma unroll
      for (int m = 0; m < 8; ++m) wx[m] = sWm[(e*8 + (q ^ m))*8 + q];
      int srcg = (lane & 16) | e;
      #pragma unroll
      for (int v = 0; v < 8; ++v) {
        float hv = accd[c][v] + bdv;
        float h2 = bmv + hv * wx[0];
        #pragma unroll
        for (int m = 1; m < 8; ++m)
          h2 += __shfl_xor(hv, m, 8) * wx[m];
        float g = __shfl(gate[v], srcg, 32);
        scr[(v + hi*8)*SCR_STRIDE + c*16 + li] = f2bf(g * h2);
      }
    }
    if (li < E_) {                       // gate columns (K rows 64..71 -> bu)
      #pragma unroll
      for (int v = 0; v < 8; ++v)
        scr[(v + hi*8)*SCR_STRIDE + 64 + li] = f2bf(gate[v]);
    }

    asm volatile("s_wait_dscnt 0" ::: "memory");  // LDS turnaround within wave

    // ---- up projection: [16 x 96] . [96 x 768], bu folded via gate rows ----
    const bf16* scr_row = scr + li * SCR_STRIDE;
    for (int nt = 0; nt < NT_UP; ++nt) {
      v8f acc = {0,0,0,0,0,0,0,0};
      #pragma unroll
      for (int kk = 0; kk < KUP; ++kk) {
        v16bf a = load_a_scratch(scr_row, kk, hi);
        acc = wmma_bf16(a, ldsB(sWu + ((nt*KUP + kk)*32 + lane)*16), acc);
      }
      int d = nt*16 + li;
      float* op = out + (size_t)(m0 + hi*8)*D_ + d;
      #pragma unroll
      for (int v = 0; v < 8; ++v) op[(size_t)v * D_] = acc[v];
    }
    asm volatile("s_wait_dscnt 0" ::: "memory");  // scratch reuse next tile
  }
}

extern "C" void kernel_launch(void* const* d_in, const int* in_sizes, int n_in,
                              void* d_out, int out_size, void* d_ws, size_t ws_size,
                              hipStream_t stream) {
  const float* x  = (const float*)d_in[0];
  const float* Wd = (const float*)d_in[1];
  const float* bd = (const float*)d_in[2];
  const float* Wm = (const float*)d_in[3];
  const float* bm = (const float*)d_in[4];
  const float* Wu = (const float*)d_in[5];
  const float* bu = (const float*)d_in[6];
  const float* Wg = (const float*)d_in[7];
  const float* bg = (const float*)d_in[8];
  float* out = (float*)d_out;

  (void)in_sizes; (void)n_in; (void)out_size; (void)d_ws; (void)ws_size;
  static_assert(SMEM_BYTES <= 320*1024, "LDS over 320KB/WGP");
  hipFuncSetAttribute((const void*)moe_fused,
                      hipFuncAttributeMaxDynamicSharedMemorySize, SMEM_BYTES);

  // 128 persistent workgroups x 8 waves; each wave owns 16-token tiles (2 each)
  moe_fused<<<dim3(128), dim3(256), SMEM_BYTES, stream>>>(
      x, Wd, bd, Wm, bm, Wu, bu, Wg, bg, out);
}
